// SelfAttentionStateMapping_730144440723
// MI455X (gfx1250) — matrix operand order repaired and checked by reading, not compiled, not run on hardware
//
#include <hip/hip_runtime.h>
#include <hip/hip_bf16.h>

typedef __attribute__((ext_vector_type(16))) _Float16 v16h;
typedef __attribute__((ext_vector_type(8)))  float    v8f;

#define B_DIM 16
#define S_DIM 2048
#define F_DIM 128
#define E_DIM 512
#define SCORE_STRIDE 2056   // 2048 + 8 pad (keeps float4 alignment: 2056 % 8 == 0)

// ---------------------------------------------------------------------------
// Kernel 1: Q = XWq^T+bq, K = XWk^T+bk, V = XWv^T+bv  (fp32 math, f16 out)
// V is written transposed: Vt[b][f][s] so the attn@V B-matrix loads are
// contiguous along K (the sequence dim).
// ---------------------------------------------------------------------------
__global__ void qkv_kernel(const float* __restrict__ X,
                           const float* __restrict__ Wq, const float* __restrict__ bq,
                           const float* __restrict__ Wk, const float* __restrict__ bk,
                           const float* __restrict__ Wv, const float* __restrict__ bv,
                           _Float16* __restrict__ Qh, _Float16* __restrict__ Kh,
                           _Float16* __restrict__ Vt)
{
    __shared__ __align__(16) float xrow[F_DIM];
    const int tid = threadIdx.x;          // 128 threads: one per output feature
    const int rs  = blockIdx.x;           // b*S + s
    xrow[tid] = X[(size_t)rs * F_DIM + tid];
    __syncthreads();

    const int b = rs >> 11;
    const int s = rs & (S_DIM - 1);

    float q = bq[tid], k = bk[tid], v = bv[tid];
    const float4* x4  = (const float4*)xrow;
    const float4* wq4 = (const float4*)(Wq + (size_t)tid * F_DIM);
    const float4* wk4 = (const float4*)(Wk + (size_t)tid * F_DIM);
    const float4* wv4 = (const float4*)(Wv + (size_t)tid * F_DIM);
#pragma unroll 8
    for (int i = 0; i < F_DIM / 4; ++i) {
        float4 xv = x4[i];
        float4 a = wq4[i], c = wk4[i], d = wv4[i];
        q += xv.x * a.x + xv.y * a.y + xv.z * a.z + xv.w * a.w;
        k += xv.x * c.x + xv.y * c.y + xv.z * c.z + xv.w * c.w;
        v += xv.x * d.x + xv.y * d.y + xv.z * d.z + xv.w * d.w;
    }
    Qh[(size_t)rs * F_DIM + tid] = (_Float16)q;
    Kh[(size_t)rs * F_DIM + tid] = (_Float16)k;
    Vt[((size_t)b * F_DIM + tid) * S_DIM + s] = (_Float16)v;
}

// ---------------------------------------------------------------------------
// Kernel 2: fused scores -> softmax -> attn store -> ctx = attn @ V
// One 256-thread block (8 wave32) per (batch, 16-query block).
// Scores for the whole 16 x 2048 block live in LDS (128 KB < 320 KB WGP).
// ---------------------------------------------------------------------------
__global__ void attn_kernel(const _Float16* __restrict__ Qh,
                            const _Float16* __restrict__ Kh,
                            const _Float16* __restrict__ Vt,
                            _Float16* __restrict__ ctxh,
                            float* __restrict__ attn_out)
{
    extern __shared__ float scores[];                 // [16][SCORE_STRIDE]
    __shared__ __align__(16) _Float16 qtile[16 * F_DIM];
    __shared__ float red[256];
    __shared__ float rowstat[16];

    const int tid  = threadIdx.x;
    const int lane = tid & 31;
    const int wave = tid >> 5;
    const int b    = blockIdx.x >> 7;                 // 128 q-blocks per batch
    const int q0   = (blockIdx.x & 127) * 16;

    // ---- stage the 16x128 f16 Q tile into LDS (2048 halves = 256 uint4) ----
    const _Float16* Qbase = Qh + ((size_t)(b * S_DIM + q0)) * F_DIM;
    ((uint4*)qtile)[tid] = ((const uint4*)Qbase)[tid];
    __syncthreads();

    // ---- build the 4 A-fragments (16x32 f16 each) in contiguous k order (k = 16h + s) ----
    // lane<16 : row=lane,   K = koff+0..15 with koff=0
    // lane>=16: row=lane-16, koff=16
    const int arow = lane & 15;
    const int koff = (lane >> 4) << 4;                // 0 or 16
    v16h aQ[4];
#pragma unroll
    for (int kb = 0; kb < 4; ++kb) {
        const _Float16* p = &qtile[arow * F_DIM + kb * 32 + koff];
#pragma unroll
        for (int i = 0; i < 8; ++i) { aQ[kb][i] = p[i]; aQ[kb][8 + i] = p[8 + i]; }
    }

    // ---- scores: each wave sweeps key tiles kt = wave, wave+8, ... ----
    const float scale = 0.08838834764831845f;         // 1/sqrt(128)
    const int   chi   = (lane >> 4) << 3;             // C/D: rows r and r+8 split by half-wave
    for (int kt = wave; kt < S_DIM / 16; kt += 8) {
        const int keyrow = kt * 16 + (lane & 15);     // B: lane holds column N=key row
        const _Float16* kp = Kh + ((size_t)(b * S_DIM + keyrow)) * F_DIM + ((lane >> 4) << 4);
        v8f c = {};
#pragma unroll
        for (int kb = 0; kb < 4; ++kb) {
            v16h bf = *(const v16h*)(kp + kb * 32);   // 16 contiguous K-halves, 32B aligned
            c = __builtin_amdgcn_wmma_f32_16x16x32_f16(false, aQ[kb], false, bf,
                                                       (short)0, c, false, false);
        }
        const int col = kt * 16 + (lane & 15);
#pragma unroll
        for (int r = 0; r < 8; ++r)
            scores[(r + chi) * SCORE_STRIDE + col] = c[r] * scale;
    }
    __syncthreads();

    // ---- softmax over each of the 16 rows (16 threads/row, 128 cols each) ----
    const int srow = tid >> 4;
    const int seg  = tid & 15;
    float* sp = &scores[srow * SCORE_STRIDE + seg * 128];
    float m = -3.4e38f;
    for (int i = 0; i < 128; ++i) m = fmaxf(m, sp[i]);
    red[tid] = m;
    __syncthreads();
    if (seg == 0) {
        float mm = red[srow * 16];
        for (int i = 1; i < 16; ++i) mm = fmaxf(mm, red[srow * 16 + i]);
        rowstat[srow] = mm;
    }
    __syncthreads();
    const float mrow = rowstat[srow];
    float ssum = 0.f;
    for (int i = 0; i < 128; ++i) { float e = __expf(sp[i] - mrow); sp[i] = e; ssum += e; }
    red[tid] = ssum;
    __syncthreads();
    if (seg == 0) {
        float s = 0.f;
        for (int i = 0; i < 16; ++i) s += red[srow * 16 + i];
        rowstat[srow] = 1.0f / s;
    }
    __syncthreads();
    const float inv = rowstat[srow];
    // normalize in LDS and stream attn to HBM exactly once (float4 stores)
    float* arow_out = attn_out + ((size_t)(b * S_DIM) + q0 + srow) * S_DIM + seg * 128;
    for (int i = 0; i < 128; i += 4) {
        float4 v;
        v.x = sp[i + 0] * inv; v.y = sp[i + 1] * inv;
        v.z = sp[i + 2] * inv; v.w = sp[i + 3] * inv;
        sp[i + 0] = v.x; sp[i + 1] = v.y; sp[i + 2] = v.z; sp[i + 3] = v.w;
        *(float4*)(arow_out + i) = v;
    }
    __syncthreads();

    // ---- ctx = attn(16x2048) @ V(2048x128): wave w owns feature tile n0=w*16 ----
    const int n0   = wave * 16;
    const int ncol = n0 + (lane & 15);
    const _Float16* vtc = Vt + ((size_t)b * F_DIM + ncol) * S_DIM + ((lane >> 4) << 4);
    v8f c = {};
    for (int kb = 0; kb < S_DIM / 32; ++kb) {
        v16h a;
        const float* ap = &scores[(lane & 15) * SCORE_STRIDE + kb * 32 + koff];
        const float4* ap4 = (const float4*)ap;        // 32B aligned (stride 2056 = 8*257)
        float4 lo0 = ap4[0], lo1 = ap4[1];            // K = koff..koff+7
        float4 hi0 = ap4[2], hi1 = ap4[3];            // K = koff+8..koff+15
        a[0]  = (_Float16)lo0.x; a[1]  = (_Float16)lo0.y; a[2]  = (_Float16)lo0.z; a[3]  = (_Float16)lo0.w;
        a[4]  = (_Float16)lo1.x; a[5]  = (_Float16)lo1.y; a[6]  = (_Float16)lo1.z; a[7]  = (_Float16)lo1.w;
        a[8]  = (_Float16)hi0.x; a[9]  = (_Float16)hi0.y; a[10] = (_Float16)hi0.z; a[11] = (_Float16)hi0.w;
        a[12] = (_Float16)hi1.x; a[13] = (_Float16)hi1.y; a[14] = (_Float16)hi1.z; a[15] = (_Float16)hi1.w;
        v16h bf = *(const v16h*)(vtc + kb * 32);
        c = __builtin_amdgcn_wmma_f32_16x16x32_f16(false, a, false, bf,
                                                   (short)0, c, false, false);
    }
    _Float16* cb = ctxh + ((size_t)(b * S_DIM + q0)) * F_DIM;
#pragma unroll
    for (int r = 0; r < 8; ++r)
        cb[(r + chi) * F_DIM + n0 + (lane & 15)] = (_Float16)c[r];
}

// ---------------------------------------------------------------------------
// Kernel 3: out = ctx @ Wo^T + bo   (16 rows x 512 cols per 256-thread block)
// ---------------------------------------------------------------------------
__global__ void outproj_kernel(const _Float16* __restrict__ ctxh,
                               const float* __restrict__ Wo,
                               const float* __restrict__ bo,
                               float* __restrict__ out)
{
    const int tid  = threadIdx.x;
    const int lane = tid & 31;
    const int wave = tid >> 5;
    const int row0 = blockIdx.x * 16;                 // global (b*S+s) row

    const int arow = lane & 15;
    const int koff = (lane >> 4) << 4;
    const int chi  = (lane >> 4) << 3;

    // A fragments: 16 ctx rows, K=128 (f16 from workspace)
    v16h a[4];
    const _Float16* ap = ctxh + ((size_t)(row0 + arow)) * F_DIM;
#pragma unroll
    for (int kb = 0; kb < 4; ++kb) {
#pragma unroll
        for (int i = 0; i < 8; ++i) {
            a[kb][i]     = ap[kb * 32 + koff + i];
            a[kb][8 + i] = ap[kb * 32 + koff + 8 + i];
        }
    }

    // wave w covers columns [w*64, w*64+64): four 16x16 output tiles
    for (int t = 0; t < 4; ++t) {
        const int n0   = wave * 64 + t * 16;
        const int ecol = n0 + (lane & 15);
        const float* wr = Wo + (size_t)ecol * F_DIM + ((lane >> 4) << 4);
        v8f c = {};
#pragma unroll
        for (int kb = 0; kb < 4; ++kb) {
            const float4* w4 = (const float4*)(wr + kb * 32);
            float4 w0 = w4[0], w1 = w4[1], w2 = w4[2], w3 = w4[3];
            v16h bf;
            bf[0]  = (_Float16)w0.x; bf[1]  = (_Float16)w0.y; bf[2]  = (_Float16)w0.z; bf[3]  = (_Float16)w0.w;
            bf[4]  = (_Float16)w1.x; bf[5]  = (_Float16)w1.y; bf[6]  = (_Float16)w1.z; bf[7]  = (_Float16)w1.w;
            bf[8]  = (_Float16)w2.x; bf[9]  = (_Float16)w2.y; bf[10] = (_Float16)w2.z; bf[11] = (_Float16)w2.w;
            bf[12] = (_Float16)w3.x; bf[13] = (_Float16)w3.y; bf[14] = (_Float16)w3.z; bf[15] = (_Float16)w3.w;
            c = __builtin_amdgcn_wmma_f32_16x16x32_f16(false, a[kb], false, bf,
                                                       (short)0, c, false, false);
        }
        const float bias = bo[ecol];
#pragma unroll
        for (int r = 0; r < 8; ++r)
            out[((size_t)(row0 + r + chi)) * E_DIM + ecol] = c[r] + bias;
    }
}

// ---------------------------------------------------------------------------
extern "C" void kernel_launch(void* const* d_in, const int* in_sizes, int n_in,
                              void* d_out, int out_size, void* d_ws, size_t ws_size,
                              hipStream_t stream) {
    (void)in_sizes; (void)n_in; (void)out_size; (void)ws_size;
    const float* X  = (const float*)d_in[0];
    const float* Wq = (const float*)d_in[1];
    const float* bq = (const float*)d_in[2];
    const float* Wk = (const float*)d_in[3];
    const float* bk = (const float*)d_in[4];
    const float* Wv = (const float*)d_in[5];
    const float* bv = (const float*)d_in[6];
    const float* Wo = (const float*)d_in[7];
    const float* bo = (const float*)d_in[8];

    float* out  = (float*)d_out;                                   // [16,2048,512]
    float* attn = out + (size_t)B_DIM * S_DIM * E_DIM;             // [16,2048,2048]

    const size_t NQK = (size_t)B_DIM * S_DIM * F_DIM;              // 4,194,304 halves
    _Float16* Qh   = (_Float16*)d_ws;
    _Float16* Kh   = Qh + NQK;
    _Float16* Vt   = Kh + NQK;   // transposed [B][F][S]
    _Float16* ctxh = Vt + NQK;

    qkv_kernel<<<B_DIM * S_DIM, F_DIM, 0, stream>>>(X, Wq, bq, Wk, bk, Wv, bv, Qh, Kh, Vt);

    const size_t smem = (size_t)16 * SCORE_STRIDE * sizeof(float); // ~128.5 KB in-WGP
    attn_kernel<<<B_DIM * (S_DIM / 16), 256, smem, stream>>>(Qh, Kh, Vt, ctxh, attn);

    outproj_kernel<<<(B_DIM * S_DIM) / 16, 256, 0, stream>>>(ctxh, Wo, bo, out);
}
